// BiLinear_69492570849793
// MI455X (gfx1250) — matrix operand order repaired and checked by reading, not compiled, not run on hardware
//
#include <hip/hip_runtime.h>

// ---------------------------------------------------------------------------
// Fused EMLP BiLinear for MI455X (gfx1250, wave32, WMMA).
//   out[b,o] = 0.05 * sum_k E[b,k] * x[b, i_k]   with (o_k,i_k) = invperm(k)
//   E computed on the fly in 16x16 tiles via v_wmma_f32_16x16x32_f16,
//   scattered straight into an LDS accumulator (ds_add_f32). Wflat (268 MB)
//   is never materialized.
// Round-2 fixes (scatter was 8 serialized LDS round-trips per tile):
//   * x tile stored TRANSPOSED in LDS (f16): a lane's 8 batch-row values at
//     column i are contiguous -> one ds_load_b128 instead of 8 ds_load_u16.
//   * accumulator stored [o][bRow]: a lane's 8 ds_add_f32 hit 8 consecutive
//     DWORDs (8 distinct banks) instead of one bank 8 times.
//   * invp for a type-1 super-tile prefetched as 3x int4 (consecutive k's,
//     16B aligned) before the WMMA chain.
//   * type-1 WMMAs issued in groups of 4, scatter after, so matrix ops
//     overlap the previous group's LDS traffic.
// ---------------------------------------------------------------------------

typedef __attribute__((ext_vector_type(16))) _Float16 v16h;
typedef __attribute__((ext_vector_type(8)))  _Float16 v8h;
typedef __attribute__((ext_vector_type(8)))  float    v8f;

namespace {
constexpr int Bsz   = 256;
constexpr int DIN   = 512;
constexpr int DOUT  = 512;
constexpr int M0    = 22144, N0 = 128;
constexpr int M1    = 20000, N1 = 32, S1 = 12;
constexpr int WSIZE = DOUT * DIN;          // 262144
constexpr int NT0   = M0 / 16;             // 1384 type-0 tiles (16 k-cols each)
constexpr int NT1   = M1 / 16;             // 1250 type-1 SUPER-tiles (16 m x 12 s)
constexpr int NST   = NT0 + NT1;           // 2634 work items
constexpr int BT    = 16;                  // batch tile (WMMA M)
constexpr int NBT   = Bsz / BT;            // 16 batch tiles
constexpr int BPB   = 64;                  // blocks per batch tile
constexpr int THREADS = 256, NWAVES = 8;
constexpr int WPB   = BPB * NWAVES;        // 512 waves walking one batch tile
}

// B fragment (32x16 f16): lane holds K = 16*g + h for its column; those are 16
// consecutive floats of one P row -> 4 coalesced float4 loads.
__device__ __forceinline__ v16h load_fragB(const float* __restrict__ p) {
  float4 q0 = ((const float4*)p)[0];
  float4 q1 = ((const float4*)p)[1];
  float4 q2 = ((const float4*)p)[2];
  float4 q3 = ((const float4*)p)[3];
  v16h b;
  b[0]  = (_Float16)q0.x; b[1]  = (_Float16)q0.y; b[2]  = (_Float16)q0.z; b[3]  = (_Float16)q0.w;
  b[4]  = (_Float16)q1.x; b[5]  = (_Float16)q1.y; b[6]  = (_Float16)q1.z; b[7]  = (_Float16)q1.w;
  b[8]  = (_Float16)q2.x; b[9]  = (_Float16)q2.y; b[10] = (_Float16)q2.z; b[11] = (_Float16)q2.w;
  b[12] = (_Float16)q3.x; b[13] = (_Float16)q3.y; b[14] = (_Float16)q3.z; b[15] = (_Float16)q3.w;
  return b;
}

__global__ void zero_out_kernel(float* __restrict__ out, int n) {
  int i = blockIdx.x * blockDim.x + threadIdx.x;
  if (i < n) out[i] = 0.0f;
}

__global__ void build_invp_kernel(const int* __restrict__ perm,
                                  int* __restrict__ invp) {
  int j = blockIdx.x * blockDim.x + threadIdx.x;
  if (j < WSIZE) invp[perm[j]] = j;   // perm is a permutation: every slot written
}

__global__ __launch_bounds__(THREADS)
void bilinear_fused_kernel(const float* __restrict__ x,
                           const float* __restrict__ w,
                           const int*   __restrict__ bids0,
                           const int*   __restrict__ bids1,
                           const int*   __restrict__ invp,
                           float*       __restrict__ out) {
  // Transposed tiles: column index fastest over the 16 batch rows.
  __shared__ alignas(16) _Float16 xtileT[DIN * BT];  // 16 KB: xT[i][bRow]
  __shared__ alignas(16) float    acc[DOUT * BT];    // 32 KB: acc[o][bRow]
  __shared__ int sb0[N0];
  __shared__ int sb1[N1 * S1];

  const int tid   = threadIdx.x;
  const int btile = blockIdx.x / BPB;
  const int blkin = blockIdx.x % BPB;
  const int bBase = btile * BT;

  for (int idx = tid; idx < BT * DIN; idx += THREADS) {
    int r = idx >> 9, c = idx & (DIN - 1);          // coalesced global read
    xtileT[c * BT + r] = (_Float16)x[(bBase + r) * DIN + c];
    acc[idx] = 0.0f;
  }
  for (int idx = tid; idx < N0;      idx += THREADS) sb0[idx] = bids0[idx];
  for (int idx = tid; idx < N1 * S1; idx += THREADS) sb1[idx] = bids1[idx];
  __syncthreads();

  const int lane = tid & 31;
  const int wave = tid >> 5;
  const int g    = (lane >> 4) & 1;     // half-wave group
  const int mrow = lane & 15;           // A: M row / B,D: N column owner

  // Scatter one 16x16 E tile given the (pre-loaded) inverse-perm entry j.
  // D layout: lane owns col N = lane&15, rows M = v + 8*g. The 8 x values
  // needed are a contiguous half-column of xtileT -> one ds_load_b128.
  auto scatter = [&](v8f c, int j) {
    int o = j >> 9, i = j & 511;
    v8h xh = *(const v8h*)&xtileT[i * BT + 8 * g];
#pragma unroll
    for (int v = 0; v < 8; ++v) {
      float val = c[v] * (float)xh[v];
      atomicAdd(&acc[o * BT + 8 * g + v], val);     // 8 consecutive DWORDs
    }
  };

  // ---- Build A fragments once; reused for every tile this wave touches. ----
  // A 16x32 as used here (contiguous k, same as B): half h -> K = 16*g + h, M = lane&15.
  v16h a0[4];                           // Xg0 row, K-slices of n in [0,128)
#pragma unroll
  for (int kc = 0; kc < 4; ++kc) {
#pragma unroll
    for (int h = 0; h < 16; ++h) {
      int K = (g << 4) | h;
      a0[kc][h] = xtileT[sb0[kc * 32 + K] * BT + mrow];
    }
  }
  v16h a1[S1];                          // Xg1 rows per irrep slot s, n in [0,32)
#pragma unroll
  for (int s = 0; s < S1; ++s) {
#pragma unroll
    for (int h = 0; h < 16; ++h) {
      int K = (g << 4) | h;
      a1[s][h] = xtileT[sb1[K * S1 + s] * BT + mrow];
    }
  }

  // ---- Strided walk over the 2634 work items for this batch tile. ----
  const int wib = blkin * NWAVES + wave;            // wave id within batch tile
  for (int t = wib; t < NST; t += WPB) {
    if (t < NT0) {
      // E0 tile: Xg0 (16x128) @ P0^T cols -> 4 chained WMMAs over K.
      int rowBase = t * 16;
      int j = invp[rowBase + mrow];                 // overlap with WMMA chain
      const float* wrow = w + (size_t)(rowBase + mrow) * N0 + g * 16;
      v8f c = {};
#pragma unroll
      for (int kc = 0; kc < 4; ++kc) {
        v16h b = load_fragB(wrow + kc * 32);
        c = __builtin_amdgcn_wmma_f32_16x16x32_f16(false, a0[kc], false, b,
                                                   (short)0, c, false, false);
      }
      scatter(c, j);                                // k = m
    } else {
      // E1 super-tile: one B fragment (P1 rows, s-independent) feeds 12 WMMAs.
      int mt = t - NT0;
      int kb = M0 + (mt * 16 + mrow) * S1;          // k = M0 + m*12 + s
      // 12 consecutive invp entries, 16B aligned -> 3 vector loads up front.
      int4 j0 = *(const int4*)(invp + kb);
      int4 j1 = *(const int4*)(invp + kb + 4);
      int4 j2 = *(const int4*)(invp + kb + 8);
      const int jj[S1] = {j0.x, j0.y, j0.z, j0.w,
                          j1.x, j1.y, j1.z, j1.w,
                          j2.x, j2.y, j2.z, j2.w};
      const float* wrow = w + (size_t)M0 * N0
                            + (size_t)(mt * 16 + mrow) * N1 + g * 16;
      v16h b = load_fragB(wrow);
#pragma unroll
      for (int grp = 0; grp < 3; ++grp) {           // 4 WMMAs, then 4 scatters
        v8f cc[4];
#pragma unroll
        for (int u = 0; u < 4; ++u)
          cc[u] = __builtin_amdgcn_wmma_f32_16x16x32_f16(
              false, a1[grp * 4 + u], false, b, (short)0, v8f{}, false, false);
#pragma unroll
        for (int u = 0; u < 4; ++u)
          scatter(cc[u], jj[grp * 4 + u]);
      }
    }
  }
  __syncthreads();

  // Merge this block's partial sums into global out, applying 0.05.
  for (int idx = tid; idx < BT * DOUT; idx += THREADS) {
    int o = idx >> 4, r = idx & (BT - 1);
    atomicAdd(&out[(bBase + r) * DOUT + o], 0.05f * acc[idx]);
  }
}

extern "C" void kernel_launch(void* const* d_in, const int* in_sizes, int n_in,
                              void* d_out, int out_size, void* d_ws, size_t ws_size,
                              hipStream_t stream) {
  (void)in_sizes; (void)n_in; (void)out_size; (void)ws_size;
  const float* x     = (const float*)d_in[0];
  const float* w     = (const float*)d_in[1];
  const int*   bids0 = (const int*)d_in[2];
  const int*   bids1 = (const int*)d_in[3];
  const int*   perm  = (const int*)d_in[4];
  float* out  = (float*)d_out;
  int*   invp = (int*)d_ws;                         // 262144 * 4 = 1 MB scratch

  zero_out_kernel<<<(Bsz * DOUT + 255) / 256, 256, 0, stream>>>(out, Bsz * DOUT);
  build_invp_kernel<<<(WSIZE + 255) / 256, 256, 0, stream>>>(perm, invp);
  bilinear_fused_kernel<<<NBT * BPB, THREADS, 0, stream>>>(x, w, bids0, bids1,
                                                           invp, out);
}
